// SparseConvBlock_77318001262914
// MI455X (gfx1250) — matrix operand order repaired and checked by reading, not compiled, not run on hardware
//
#include <hip/hip_runtime.h>
#include <math.h>

// ---- problem constants (fixed by the reference) ----
#define NPTS 1000000
#define KT   27         // kernel taps
#define CIN  32
#define COUT 64
#define WGS  512
#define WAVES (WGS/32)                       // 16 waves
#define ROWS_PER_WG (WAVES*16)               // 256 rows per workgroup
#define NWG ((NPTS + ROWS_PER_WG - 1)/ROWS_PER_WG)   // 3907
#define KCH  7                               // taps per staging phase (conv2)
#define NPH  ((KT + KCH - 1)/KCH)            // 4 phases

typedef _Float16 v16h __attribute__((ext_vector_type(16)));
typedef float    v8f  __attribute__((ext_vector_type(8)));

union H16 { v16h v; _Float16 h[16]; };
union F8  { v8f  v; float f[8]; };
union Q8  { uint4 q; _Float16 h[8]; };
union FQ  { float4 q; float f[4]; };

// =====================================================================
// Layer 1: gather(feats) -> f16 WMMA -> raw f32 accum -> f16 h1 + BN partials
// Software-pipelined: tap k+1's gather is issued before tap k's WMMAs.
// =====================================================================
__global__ __launch_bounds__(WGS, 2)
void conv1_kernel(const float* __restrict__ feats,
                  const int* __restrict__ nbr,
                  const unsigned char* __restrict__ maskb,
                  const float* __restrict__ W1,
                  _Float16* __restrict__ h1,
                  float* __restrict__ partials)
{
    __shared__ __align__(32) _Float16 sw[KT*4*32*16];     // W1 fragments (110592 B)
    __shared__ int           sidx[WAVES][16*KT];
    __shared__ unsigned char smsk[WAVES][16*KT];
    __shared__ float         psum[WAVES][COUT];
    __shared__ float         psq [WAVES][COUT];

    const int tid  = threadIdx.x;
    const int wave = tid >> 5;
    const int lane = tid & 31;
    const int lh   = lane >> 4;      // lane half: which K-subset of the A row
    const int m    = lane & 15;      // row within the 16-row tile
    const int row_base = blockIdx.x * ROWS_PER_WG + wave * 16;
    const bool active  = row_base < NPTS;

    // ---- stage W1 (f32 -> f16) into WMMA B-fragment layout ----
    for (int e = tid; e < KT*CIN*COUT; e += WGS) {
        int k  = e / (CIN*COUT);
        int r  = e - k*(CIN*COUT);
        int kk = r >> 6;
        int n  = r & 63;
        int nt = n >> 4;
        int ln = (((kk >> 3) & 1) << 4) | (n & 15);
        int ps = (kk & 7) | ((kk >> 4) << 3);
        sw[(((k*4)+nt)*32 + ln)*16 + ps] = (_Float16)W1[e];
    }
    if (active) {
        const long gb = (long)row_base * KT;
        for (int t = lane; t < 16*KT; t += 32) {
            sidx[wave][t] = nbr[gb + t];
            smsk[wave][t] = maskb[gb + t];
        }
    }
    __syncthreads();

    v8f cacc[4] = {};
    if (active) {
        FQ buf[2][4];
        int mkc = smsk[wave][m*KT];
        {
            const int idx0 = sidx[wave][m*KT];
            if (mkc) {
                const float4* p = (const float4*)(feats + (size_t)idx0 * CIN);
                buf[0][0].q = p[2*lh];     buf[0][1].q = p[2*lh + 1];
                buf[0][2].q = p[4 + 2*lh]; buf[0][3].q = p[5 + 2*lh];
            }
        }
        #pragma unroll 2
        for (int k = 0; k < KT; ++k) {
            const int cur = k & 1, nxt = cur ^ 1;
            int mkn = 0;
            if (k + 1 < KT) {                       // prefetch next tap's row
                mkn = smsk[wave][m*KT + k + 1];
                const int idxn = sidx[wave][m*KT + k + 1];
                if (mkn) {
                    const float4* p = (const float4*)(feats + (size_t)idxn * CIN);
                    buf[nxt][0].q = p[2*lh];     buf[nxt][1].q = p[2*lh + 1];
                    buf[nxt][2].q = p[4 + 2*lh]; buf[nxt][3].q = p[5 + 2*lh];
                }
            }
            H16 a;
            if (mkc) {
                #pragma unroll
                for (int j = 0; j < 4; ++j) {
                    a.h[j]      = (_Float16)buf[cur][0].f[j];
                    a.h[4 + j]  = (_Float16)buf[cur][1].f[j];
                    a.h[8 + j]  = (_Float16)buf[cur][2].f[j];
                    a.h[12 + j] = (_Float16)buf[cur][3].f[j];
                }
            } else {
                #pragma unroll
                for (int j = 0; j < 16; ++j) a.h[j] = (_Float16)0.f;
            }
            #pragma unroll
            for (int nt = 0; nt < 4; ++nt) {
                v16h b = *(const v16h*)&sw[(((k*4)+nt)*32 + lane)*16];
                cacc[nt] = __builtin_amdgcn_wmma_f32_16x16x32_f16(
                    false, a.v, false, b, (short)0, cacc[nt], false, false);
            }
            mkc = mkn;
        }
        // ---- store raw h1 as f16, accumulate BN partial sums ----
        #pragma unroll
        for (int nt = 0; nt < 4; ++nt) {
            F8 c; c.v = cacc[nt];
            float s = 0.f, q = 0.f;
            #pragma unroll
            for (int r = 0; r < 8; ++r) {
                float v = c.f[r];
                int mr = 8*lh + r;
                h1[(size_t)(row_base + mr)*COUT + nt*16 + m] = (_Float16)v;
                s += v; q += v*v;
            }
            s += __shfl_xor(s, 16);
            q += __shfl_xor(q, 16);
            if (lh == 0) { psum[wave][nt*16 + m] = s; psq[wave][nt*16 + m] = q; }
        }
    } else {
        psum[wave][lane] = 0.f; psum[wave][32 + lane] = 0.f;
        psq [wave][lane] = 0.f; psq [wave][32 + lane] = 0.f;
    }
    __syncthreads();
    if (tid < 128) {
        int c = tid & 63;
        float acc = 0.f;
        if (tid < 64) { for (int w = 0; w < WAVES; ++w) acc += psum[w][c]; }
        else          { for (int w = 0; w < WAVES; ++w) acc += psq [w][c]; }
        partials[(size_t)blockIdx.x * 128 + tid] = acc;
    }
}

// =====================================================================
// BN reduce: deterministic double-precision sum over per-WG partials,
// emit per-channel (a, b): y = a*x + b
// =====================================================================
__global__ void reduce_bn_kernel(const float* __restrict__ partials,
                                 const float* __restrict__ gamma,
                                 const float* __restrict__ beta,
                                 float* __restrict__ coeff)   // a[64], b[64]
{
    __shared__ double acc[128];
    const int t = threadIdx.x;
    double a = 0.0;
    for (int w = 0; w < NWG; ++w) a += (double)partials[(size_t)w * 128 + t];
    acc[t] = a;
    __syncthreads();
    if (t < 64) {
        const double inv = 1.0 / (double)NPTS;
        double mu  = acc[t] * inv;
        double var = acc[64 + t] * inv - mu * mu;
        float scale = (float)((double)gamma[t] / sqrt(var + 1e-5));
        coeff[t]      = scale;
        coeff[64 + t] = beta[t] - (float)mu * scale;
    }
}

// =====================================================================
// Layer 2: gather(h1 f16) -> fused BN1+ReLU -> WMMA -> raw f32 out + BN partials
// W2 staged in 4 phases of 7 taps (57 KB LDS) so 2-3 WGs fit per WGP;
// accumulators persist in VGPRs across phases. Gathers double-buffered.
// =====================================================================
__global__ __launch_bounds__(WGS, 2)
void conv2_kernel(const _Float16* __restrict__ h1,
                  const int* __restrict__ nbr,
                  const unsigned char* __restrict__ maskb,
                  const float* __restrict__ W2,
                  const float* __restrict__ coeff,   // a1[64], b1[64]
                  float* __restrict__ out,
                  float* __restrict__ partials)
{
    __shared__ __align__(32) _Float16 sw[KCH*2*4*32*16];  // 57344 B weight window
    __shared__ int           sidx[WAVES][16*KT];
    __shared__ unsigned char smsk[WAVES][16*KT];
    __shared__ float         psum[WAVES][COUT];
    __shared__ float         psq [WAVES][COUT];

    const int tid  = threadIdx.x;
    const int wave = tid >> 5;
    const int lane = tid & 31;
    const int lh   = lane >> 4;
    const int m    = lane & 15;
    const int row_base = blockIdx.x * ROWS_PER_WG + wave * 16;
    const bool active  = row_base < NPTS;

    if (active) {
        const long gb = (long)row_base * KT;
        for (int t = lane; t < 16*KT; t += 32) {
            sidx[wave][t] = nbr[gb + t];
            smsk[wave][t] = maskb[gb + t];
        }
    }

    // per-lane BN1 coefficients for the channels this lane gathers
    _Float16 ca[2][2][8], cb[2][2][8];
    #pragma unroll
    for (int s = 0; s < 2; ++s)
        #pragma unroll
        for (int w = 0; w < 2; ++w) {
            int ch0 = 32*s + 16*w + 8*lh;
            #pragma unroll
            for (int j = 0; j < 8; ++j) {
                ca[s][w][j] = (_Float16)coeff[ch0 + j];
                cb[s][w][j] = (_Float16)coeff[64 + ch0 + j];
            }
        }

    v8f cacc[4] = {};
    for (int ph = 0; ph < NPH; ++ph) {
        const int k0 = ph * KCH;
        const int k1 = (k0 + KCH < KT) ? (k0 + KCH) : KT;
        __syncthreads();                       // sw free (prev phase done)
        // ---- stage this phase's W2 window into fragment layout ----
        for (int e = tid; e < KCH*COUT*COUT; e += WGS) {
            int kl = e >> 12;                  // local tap
            int k  = k0 + kl;
            if (k < KT) {
                int r  = e & 4095;
                int kk = r >> 6;
                int n  = r & 63;
                int s  = kk >> 5;
                int kp = kk & 31;
                int nt = n >> 4;
                int ln = (((kp >> 3) & 1) << 4) | (n & 15);
                int ps = (kp & 7) | ((kp >> 4) << 3);
                sw[((((kl*2 + s)*4) + nt)*32 + ln)*16 + ps] = (_Float16)W2[(size_t)k*4096 + r];
            }
        }
        __syncthreads();

        if (active) {
            Q8 buf[2][4];
            int mkc = smsk[wave][m*KT + k0];
            {
                const int idx0 = sidx[wave][m*KT + k0];
                if (mkc) {
                    const uint4* p = (const uint4*)(h1 + (size_t)idx0 * COUT);
                    buf[0][0].q = p[lh];     buf[0][1].q = p[2 + lh];
                    buf[0][2].q = p[4 + lh]; buf[0][3].q = p[6 + lh];
                }
            }
            #pragma unroll 2
            for (int k = k0; k < k1; ++k) {
                const int cur = (k - k0) & 1, nxt = cur ^ 1;
                int mkn = 0;
                if (k + 1 < k1) {              // prefetch next tap's row
                    mkn = smsk[wave][m*KT + k + 1];
                    const int idxn = sidx[wave][m*KT + k + 1];
                    if (mkn) {
                        const uint4* p = (const uint4*)(h1 + (size_t)idxn * COUT);
                        buf[nxt][0].q = p[lh];     buf[nxt][1].q = p[2 + lh];
                        buf[nxt][2].q = p[4 + lh]; buf[nxt][3].q = p[6 + lh];
                    }
                }
                const int kl = k - k0;
                #pragma unroll
                for (int s = 0; s < 2; ++s) {
                    H16 a;
                    if (mkc) {
                        #pragma unroll
                        for (int j = 0; j < 8; ++j) {
                            _Float16 y0 = buf[cur][2*s    ].h[j]*ca[s][0][j] + cb[s][0][j];
                            _Float16 y1 = buf[cur][2*s + 1].h[j]*ca[s][1][j] + cb[s][1][j];
                            a.h[j]     = (y0 > (_Float16)0.f) ? y0 : (_Float16)0.f;
                            a.h[8 + j] = (y1 > (_Float16)0.f) ? y1 : (_Float16)0.f;
                        }
                    } else {
                        #pragma unroll
                        for (int j = 0; j < 16; ++j) a.h[j] = (_Float16)0.f;
                    }
                    #pragma unroll
                    for (int nt = 0; nt < 4; ++nt) {
                        v16h b = *(const v16h*)&sw[((((kl*2 + s)*4) + nt)*32 + lane)*16];
                        cacc[nt] = __builtin_amdgcn_wmma_f32_16x16x32_f16(
                            false, a.v, false, b, (short)0, cacc[nt], false, false);
                    }
                }
                mkc = mkn;
            }
        }
    }

    if (active) {
        #pragma unroll
        for (int nt = 0; nt < 4; ++nt) {
            F8 c; c.v = cacc[nt];
            float s = 0.f, q = 0.f;
            #pragma unroll
            for (int r = 0; r < 8; ++r) {
                float v = c.f[r];
                int mr = 8*lh + r;
                out[(size_t)(row_base + mr)*COUT + nt*16 + m] = v;
                s += v; q += v*v;
            }
            s += __shfl_xor(s, 16);
            q += __shfl_xor(q, 16);
            if (lh == 0) { psum[wave][nt*16 + m] = s; psq[wave][nt*16 + m] = q; }
        }
    } else {
        psum[wave][lane] = 0.f; psum[wave][32 + lane] = 0.f;
        psq [wave][lane] = 0.f; psq [wave][32 + lane] = 0.f;
    }
    __syncthreads();
    if (tid < 128) {
        int c = tid & 63;
        float acc = 0.f;
        if (tid < 64) { for (int w = 0; w < WAVES; ++w) acc += psum[w][c]; }
        else          { for (int w = 0; w < WAVES; ++w) acc += psq [w][c]; }
        partials[(size_t)blockIdx.x * 128 + tid] = acc;
    }
}

// =====================================================================
// Final elementwise BN2 + ReLU, in place on d_out (float4 vectorized)
// =====================================================================
__global__ void bnrelu_out_kernel(float* __restrict__ out,
                                  const float* __restrict__ coeff)  // a2[64], b2[64]
{
    size_t i = (size_t)blockIdx.x * blockDim.x + threadIdx.x;
    const size_t n4 = (size_t)NPTS * COUT / 4;
    if (i >= n4) return;
    float4* o4 = (float4*)out;
    FQ x; x.q = o4[i];
    int c = (int)((i * 4) & 63);
    #pragma unroll
    for (int j = 0; j < 4; ++j) {
        float y = x.f[j] * coeff[c + j] + coeff[64 + c + j];
        x.f[j] = y > 0.f ? y : 0.f;
    }
    o4[i] = x.q;
}

// =====================================================================
extern "C" void kernel_launch(void* const* d_in, const int* in_sizes, int n_in,
                              void* d_out, int out_size, void* d_ws, size_t ws_size,
                              hipStream_t stream) {
    const float* feats = (const float*)d_in[0];
    const float* W1    = (const float*)d_in[1];
    const float* W2    = (const float*)d_in[2];
    const float* g1    = (const float*)d_in[3];
    const float* b1    = (const float*)d_in[4];
    const float* g2    = (const float*)d_in[5];
    const float* b2    = (const float*)d_in[6];
    const int*   nbr   = (const int*)d_in[7];
    const unsigned char* mask = (const unsigned char*)d_in[8];  // jnp bool_ = 1 byte
    float* out = (float*)d_out;

    // workspace layout: h1 (f16, 128 MB) | per-WG partials (2 MB, reused) | coeffs (1 KB)
    char* ws = (char*)d_ws;
    _Float16* h1 = (_Float16*)ws;
    float* partials = (float*)(ws + (size_t)NPTS * COUT * sizeof(_Float16));
    float* coeff    = (float*)(ws + (size_t)NPTS * COUT * sizeof(_Float16)
                                  + (size_t)NWG * 128 * sizeof(float));

    conv1_kernel<<<NWG, WGS, 0, stream>>>(feats, nbr, mask, W1, h1, partials);
    reduce_bn_kernel<<<1, 128, 0, stream>>>(partials, g1, b1, coeff);
    conv2_kernel<<<NWG, WGS, 0, stream>>>(h1, nbr, mask, W2, coeff, out, partials);
    reduce_bn_kernel<<<1, 128, 0, stream>>>(partials, g2, b2, coeff + 128);
    const int n4 = NPTS * COUT / 4;
    bnrelu_out_kernel<<<(n4 + 255) / 256, 256, 0, stream>>>(out, coeff + 128);
}
